// CombinedActorModel_13752485282362
// MI455X (gfx1250) — compile-verified, matrix operand order checked by reading, not run on hardware
//
#include <hip/hip_runtime.h>
#include <hip/hip_bf16.h>

typedef __bf16   v16bf __attribute__((ext_vector_type(16)));
typedef __bf16   bf2   __attribute__((ext_vector_type(2)));
typedef float    v8f   __attribute__((ext_vector_type(8)));
typedef unsigned uvec8 __attribute__((ext_vector_type(8)));

#define THREADS 128
#define WAVES   4
#define NFRAG   19

static __device__ inline unsigned pack_bf16(float a, float b) {
    bf2 t;
    t[0] = (__bf16)a;
    t[1] = (__bf16)b;
    return __builtin_bit_cast(unsigned, t);
}

struct Frag32 { uint4 lo, hi; };

static __device__ inline v8f wmma_bf16(v16bf a, v16bf b, v8f c) {
    // (neg_a, A, neg_b, B, c_mod, C, reuse_a, reuse_b)
    return __builtin_amdgcn_wmma_f32_16x16x32_bf16(false, a, false, b, (short)0, c,
                                                   false, false);
}

// Block-sparse "big" stage-1 weights over concatenated 27-wide input.
// feature f = a*25 + j ; j 0..9 -> x(size 10), 10..19 -> y(10), 20..24 -> z(5)
// m uses input cols: x:[0,6) y:[9,15) z:[18,24) ; n uses x:[6,9) y:[15,18) z:[24,27)
// K slot 27 (stage 1) / 25 (stage 2) carries the bias (A supplies 1.0 there).
static __device__ float wm_big(const float* Wmx, const float* Wmy, const float* Wmz,
                               int f, int k) {
    int a = f / 25, j = f % 25;
    if (j < 10)      { if (k >= 0  && k < 6)  return Wmx[(a*10 + j     )*6 + k     ]; }
    else if (j < 20) { if (k >= 9  && k < 15) return Wmy[(a*10 + (j-10))*6 + (k-9) ]; }
    else             { if (k >= 18 && k < 24) return Wmz[(a*5  + (j-20))*6 + (k-18)]; }
    return 0.f;
}
static __device__ float wn_big(const float* Wnx, const float* Wny, const float* Wnz,
                               int f, int k) {
    int a = f / 25, j = f % 25;
    if (j < 10)      { if (k >= 6  && k < 9)  return Wnx[(a*10 + j     )*3 + (k-6) ]; }
    else if (j < 20) { if (k >= 15 && k < 18) return Wny[(a*10 + (j-10))*3 + (k-15)]; }
    else             { if (k >= 24 && k < 27) return Wnz[(a*5  + (j-20))*3 + (k-24)]; }
    return 0.f;
}
static __device__ float bias_big(const float* bx, const float* by, const float* bz, int f) {
    int a = f / 25, j = f % 25;
    if (j < 10) return bx[a*10 + j];
    if (j < 20) return by[a*10 + (j - 10)];
    return bz[a*5 + (j - 20)];
}

// --------- kernel 1: build pre-swizzled bf16 B-fragment table in d_ws ---------
// Table layout: dword index = frag*256 + laneSlot*8 + vgprSlot  (19 KB total)
//   frags 0..4  : WmBig n-tiles (N=75 -> 5 tiles of 16, K=27, bias @ k=27)
//   frags 5..9  : WnBig n-tiles (bias @ k=27)
//   frags 10..15: Wlin per actor (2 tiles: N=25, K=25, bias @ k=25)
//   frags 16..18: Wout per actor (N=15, 1 tile, K=25, bias @ k=25)
__global__ __launch_bounds__(256)
void build_frag_kernel(const float* __restrict__ Wmx, const float* __restrict__ bmx,
                       const float* __restrict__ Wnx, const float* __restrict__ bnx,
                       const float* __restrict__ Wmy, const float* __restrict__ bmy,
                       const float* __restrict__ Wny, const float* __restrict__ bny,
                       const float* __restrict__ Wmz, const float* __restrict__ bmz,
                       const float* __restrict__ Wnz, const float* __restrict__ bnz,
                       const float* __restrict__ Wlin, const float* __restrict__ blin,
                       const float* __restrict__ Wout, const float* __restrict__ bout,
                       unsigned* __restrict__ ws)
{
    int idx = blockIdx.x * 256 + threadIdx.x;   // [0, 19*256)
    int f   = idx >> 8;
    int rem = idx & 255;
    int l   = rem >> 3;    // lane slot
    int j   = rem & 7;     // dword (VGPR) slot
    int n   = l & 15;
    int kh  = (l >> 4) * 8;
    int k0  = ((j < 4) ? 2*j : 2*j + 8) + kh;   // K of low bf16; k0+1 in high
    float v0 = 0.f, v1 = 0.f;
    if (f < 5) {
        int col = f*16 + n;
        if (col < 75) {
            v0 = (k0   == 27) ? bias_big(bmx, bmy, bmz, col) : wm_big(Wmx, Wmy, Wmz, col, k0);
            v1 = (k0+1 == 27) ? bias_big(bmx, bmy, bmz, col) : wm_big(Wmx, Wmy, Wmz, col, k0+1);
        }
    } else if (f < 10) {
        int col = (f - 5)*16 + n;
        if (col < 75) {
            v0 = (k0   == 27) ? bias_big(bnx, bny, bnz, col) : wn_big(Wnx, Wny, Wnz, col, k0);
            v1 = (k0+1 == 27) ? bias_big(bnx, bny, bnz, col) : wn_big(Wnx, Wny, Wnz, col, k0+1);
        }
    } else if (f < 16) {
        int a = (f - 10) >> 1, t = (f - 10) & 1;
        int o = t*16 + n;
        if (o < 25) {
            if (k0     < 25)       v0 = Wlin[(a*25 + o)*25 + k0];
            else if (k0     == 25) v0 = blin[a*25 + o];
            if (k0 + 1 < 25)       v1 = Wlin[(a*25 + o)*25 + k0 + 1];
            else if (k0 + 1 == 25) v1 = blin[a*25 + o];
        }
    } else {
        int a = f - 16;
        if (n < 15) {
            if (k0     < 25)       v0 = Wout[(a*15 + n)*25 + k0];
            else if (k0     == 25) v0 = bout[a*15 + n];
            if (k0 + 1 < 25)       v1 = Wout[(a*15 + n)*25 + k0 + 1];
            else if (k0 + 1 == 25) v1 = bout[a*15 + n];
        }
    }
    ws[idx] = pack_bf16(v0, v1);
}

// --------- kernel 2: main fused model ---------
__global__ __launch_bounds__(THREADS)
void actor_ens_kernel(const float* __restrict__ spatial,
                      const unsigned* __restrict__ ws,
                      float* __restrict__ out, int Btot, int numTiles)
{
    // staging: row stride 92 / 36 (conflict-free, 16B aligned); per-actor 28-col sections
    __shared__ __attribute__((aligned(16))) float s_ps[WAVES][16][92];
    __shared__ __attribute__((aligned(16))) float s_h [WAVES][16][36];

    const int tid   = threadIdx.x;
    const int wv    = tid >> 5;
    const int lane  = tid & 31;
    const int ncol  = lane & 15;
    const int nhalf = lane >> 4;

    // load all 19 loop-invariant B fragments into registers (32 B per lane each)
    v16bf bfr[NFRAG];
    {
        const uint4* base = (const uint4*)ws + lane*2;  // frag stride = 64 uint4
        #pragma unroll
        for (int f = 0; f < NFRAG; ++f) {
            Frag32 p;
            p.lo = base[f*64 + 0];
            p.hi = base[f*64 + 1];
            bfr[f] = __builtin_bit_cast(v16bf, p);
        }
    }

    // zero this wave's staging so pad columns can't inject NaN/Inf into WMMA
    float (*ps)[92] = s_ps[wv];
    float (*hb)[36] = s_h[wv];
    { float* z = &ps[0][0]; for (int i = lane; i < 16*92; i += 32) z[i] = 0.f; }
    { float* z = &hb[0][0]; for (int i = lane; i < 16*36; i += 32) z[i] = 0.f; }
    // per-wave private buffers + in-order LDS: no __syncthreads needed

    for (int tile = blockIdx.x * WAVES + wv; tile < numTiles; tile += gridDim.x * WAVES) {
        const int rowBase = tile * 16;

        // ---- stage-1 A fragment: 16 rows x 27 of spatial (bf16), 1.0 @ k=27 ----
        v16bf aS;
        {
            int ar = rowBase + ncol;
            ar = (ar < Btot) ? ar : (Btot - 1);
            const float* srow = spatial + (size_t)ar * 27;
            uvec8 up;
            #pragma unroll
            for (int j = 0; j < 8; ++j) {
                int k = ((j < 4) ? 2*j : 2*j + 8) + nhalf*8;
                // branch-free: clamp addresses into range, select value afterwards
                int kc0 = (k     < 27) ? k     : 26;
                int kc1 = (k + 1 < 27) ? k + 1 : 26;
                float g0 = srow[kc0];
                float g1 = srow[kc1];
                float f0 = (k     < 27) ? g0 : 0.f;
                float f1 = (k + 1 < 27) ? g1 : ((k + 1 == 27) ? 1.0f : 0.f);
                up[j] = pack_bf16(f0, f1);
            }
            aS = __builtin_bit_cast(v16bf, up);
        }

        // ---- stage 1: ps = (s@WmBig^T + bm) * (s@WnBig^T + bn), 75 features ----
        #pragma unroll
        for (int t = 0; t < 5; ++t) {
            v8f cz = {0.f,0.f,0.f,0.f,0.f,0.f,0.f,0.f};
            v8f mD = wmma_bf16(aS, bfr[t], cz);
            v8f nD = wmma_bf16(aS, bfr[5 + t], cz);
            int fcol = t*16 + ncol;
            if (fcol < 75) {
                int pcol = (fcol / 25) * 28 + (fcol % 25);
                #pragma unroll
                for (int r = 0; r < 8; ++r)
                    ps[nhalf*8 + r][pcol] = mD[r] * nD[r];
            }
        }

        // ---- stage 2 per actor: lin(25x25)+softsign, out(15x25) ----
        v8f oAcc[3];
        #pragma unroll
        for (int a = 0; a < 3; ++a) {
            v16bf aP;
            {
                const float* prow = ps[ncol] + a*28;
                uvec8 up;
                #pragma unroll
                for (int j = 0; j < 8; ++j) {
                    int krel = ((j < 4) ? 2*j : 2*j + 8) + nhalf*8;
                    float f0 = prow[krel];
                    float f1 = prow[krel + 1];
                    if (j == 4) f1 = nhalf ? 1.0f : f1;   // krel==24 -> k25 bias lane
                    up[j] = pack_bf16(f0, f1);
                }
                aP = __builtin_bit_cast(v16bf, up);
            }
            #pragma unroll
            for (int t = 0; t < 2; ++t) {
                v8f cz = {0.f,0.f,0.f,0.f,0.f,0.f,0.f,0.f};
                v8f hv = wmma_bf16(aP, bfr[10 + a*2 + t], cz);
                int oc = t*16 + ncol;
                if (oc < 25) {
                    #pragma unroll
                    for (int r = 0; r < 8; ++r) {
                        float x = hv[r];
                        hb[nhalf*8 + r][oc] = __fdividef(x, 1.f + fabsf(x));  // Softsign
                    }
                }
            }
            v16bf aH;
            {
                const float* hrow = hb[ncol];
                uvec8 up;
                #pragma unroll
                for (int j = 0; j < 8; ++j) {
                    int krel = ((j < 4) ? 2*j : 2*j + 8) + nhalf*8;
                    float f0 = hrow[krel];
                    float f1 = hrow[krel + 1];
                    if (j == 4) f1 = nhalf ? 1.0f : f1;   // k25 bias lane
                    up[j] = pack_bf16(f0, f1);
                }
                aH = __builtin_bit_cast(v16bf, up);
            }
            v8f cz = {0.f,0.f,0.f,0.f,0.f,0.f,0.f,0.f};
            oAcc[a] = wmma_bf16(aH, bfr[16 + a], cz);
        }

        // ---- softmax over actors (logit = col 9), weighted sum of cols 0..8 ----
        #pragma unroll
        for (int r = 0; r < 8; ++r) {
            float v0 = oAcc[0][r], v1 = oAcc[1][r], v2 = oAcc[2][r];
            float l0 = __shfl(v0, 9, 16);   // lanes 0-15 from lane 9, 16-31 from lane 25
            float l1 = __shfl(v1, 9, 16);
            float l2 = __shfl(v2, 9, 16);
            float mx = fmaxf(l0, fmaxf(l1, l2));
            float e0 = __expf(l0 - mx);
            float e1 = __expf(l1 - mx);
            float e2 = __expf(l2 - mx);
            float res = __fdividef(v0*e0 + v1*e1 + v2*e2, e0 + e1 + e2);
            int row = rowBase + nhalf*8 + r;
            if (ncol < 9 && row < Btot)
                out[(size_t)row * 9 + ncol] = res;
        }
    }
}

extern "C" void kernel_launch(void* const* d_in, const int* in_sizes, int n_in,
                              void* d_out, int out_size, void* d_ws, size_t ws_size,
                              hipStream_t stream) {
    (void)n_in; (void)out_size; (void)ws_size;
    const float* spatial = (const float*)d_in[0];
    // d_in[1] car_stats: unused by the model
    const float* Wmx  = (const float*)d_in[2];
    const float* bmx  = (const float*)d_in[3];
    const float* Wnx  = (const float*)d_in[4];
    const float* bnx  = (const float*)d_in[5];
    const float* Wmy  = (const float*)d_in[6];
    const float* bmy  = (const float*)d_in[7];
    const float* Wny  = (const float*)d_in[8];
    const float* bny  = (const float*)d_in[9];
    const float* Wmz  = (const float*)d_in[10];
    const float* bmz  = (const float*)d_in[11];
    const float* Wnz  = (const float*)d_in[12];
    const float* bnz  = (const float*)d_in[13];
    const float* Wlin = (const float*)d_in[14];
    const float* blin = (const float*)d_in[15];
    const float* Wout = (const float*)d_in[16];
    const float* bout = (const float*)d_in[17];
    float* out = (float*)d_out;
    unsigned* ws = (unsigned*)d_ws;

    int Btot = in_sizes[0] / 27;           // spatial is [B, 3, 9]
    int numTiles = (Btot + 15) / 16;

    hipLaunchKernelGGL(build_frag_kernel, dim3(NFRAG), dim3(256), 0, stream,
                       Wmx, bmx, Wnx, bnx,
                       Wmy, bmy, Wny, bny,
                       Wmz, bmz, Wnz, bnz,
                       Wlin, blin, Wout, bout, ws);

    int blocks = (numTiles + WAVES - 1) / WAVES;
    if (blocks > 4096) blocks = 4096;
    if (blocks < 1) blocks = 1;
    hipLaunchKernelGGL(actor_ens_kernel, dim3(blocks), dim3(THREADS), 0, stream,
                       spatial, ws, out, Btot, numTiles);
}
